// control_solver_60936995995924
// MI455X (gfx1250) — matrix operand order repaired and checked by reading, not compiled, not run on hardware
//
#include <hip/hip_runtime.h>

typedef __attribute__((ext_vector_type(16))) _Float16 v16h;
typedef __attribute__((ext_vector_type(8)))  float    v8f;

// Problem constants
#define BATCH 4096
#define DIM   100
#define T_    101
#define WIDTH 200

// Padded tiling constants (WMMA 16x16x32 f16)
#define HIDP  224     // hidden width padded: 7 k-tiles of 32, 14 n-tiles of 16
#define INP   128     // input width padded (101 -> 128): 4 k-tiles
#define OUTP  128     // output width padded (100 -> 128): 8 n-tiles
#define MTILE 32      // batch rows per block (2 m-tiles of 16)
#define NWAVE 4
#define BLOCK (NWAVE * 32)

// Packed weight sizes (in f16 elements): ktiles*ntiles*512
#define SZ_W0 (4 * 14 * 512)   // 28672
#define SZ_WH (7 * 14 * 512)   // 50176
#define SZ_WO (7 * 8  * 512)   // 28672
#define TOTAL_W (SZ_W0 + 3 * SZ_WH + SZ_WO)   // 207872 halves

// ---------------------------------------------------------------------------
// Pack f32 weights (nin x nout, row-major) into fragment-major f16 layout
// matching the CDNA5 WMMA B-matrix (K x N, 32x16 per tile) VGPR layout:
//   block (kt,nt) -> 512 contiguous halves; within: lane*16 + (2v+hh)
//   element = W[kt*32 + (v>>2)*16 + (lane>>4)*8 + 2*(v&3) + hh][nt*16 + (lane&15)]
// ---------------------------------------------------------------------------
__global__ void pack_weights(const float* __restrict__ W, _Float16* __restrict__ dst,
                             int nin, int nout, int ktiles, int ntiles) {
    int total = ktiles * ntiles * 512;
    for (int idx = blockIdx.x * blockDim.x + threadIdx.x; idx < total;
         idx += gridDim.x * blockDim.x) {
        int blk    = idx >> 9;
        int within = idx & 511;
        int lane   = within >> 4;
        int hi     = within & 15;
        int v  = hi >> 1;
        int hh = hi & 1;
        int kt = blk / ntiles;
        int nt = blk - kt * ntiles;
        int k = kt * 32 + ((v >> 2) << 4) + ((lane >> 4) << 3) + ((v & 3) << 1) + hh;
        int n = (nt << 4) + (lane & 15);
        float val = (k < nin && n < nout) ? W[k * nout + n] : 0.0f;
        dst[idx] = (_Float16)val;
    }
}

// Zero-padded bias copy: removes divergent guards in the GEMM epilogues.
__global__ void pack_bias(const float* __restrict__ src, float* __restrict__ dst,
                          int nreal, int npad) {
    int i = blockIdx.x * blockDim.x + threadIdx.x;
    if (i < npad) dst[i] = (i < nreal) ? src[i] : 0.0f;
}

// ---------------------------------------------------------------------------
// A-fragment load from LDS given a per-lane row base (dword ptr) and kt.
// ISA 16-bit A layout: lane m = lane&15; VGPR v holds K = (v>=4?16:0)
// + (lane>=16?8:0) + (v&3)*2, +1. All offsets are compile-time immediates.
// ---------------------------------------------------------------------------
__device__ __forceinline__ v16h load_a_frag(const unsigned* arow, int kt, int g4) {
    union { unsigned u[8]; v16h h; } r;
#pragma unroll
    for (int v = 0; v < 8; ++v) {
        // dword index within the 16-dword (32-half) k-chunk
        int kidx = ((v >> 2) << 3) | (v & 3);
        r.u[v] = arow[kt * 16 + g4 + kidx];
    }
    return r.h;
}

// ---------------------------------------------------------------------------
// Hidden layer: Aout = relu(Ain @ W + b). Each wave owns one m-tile and 7
// n-tiles processed together: one A fragment per k-step feeds 7 independent
// WMMA accumulator chains. kt loop kept ROLLED to bound register pressure
// (7 acc + 1 A + 7 B fragments ~= 140 VGPRs, no spills, no VGPR-MSB mode).
// ---------------------------------------------------------------------------
__device__ __forceinline__ void layer_hidden(const _Float16* Ain, _Float16* Aout,
                                             const _Float16* __restrict__ Wp,
                                             int ktiles, const float* __restrict__ biasp,
                                             int lane, int wave) {
    const int ntiles = 14;               // HIDP / 16
    const int mt  = wave >> 1;           // waves 0,1 -> mt 0 ; waves 2,3 -> mt 1
    const int nt0 = (wave & 1) * 7;      // 7 n-tiles per wave
    const int g4  = (lane >> 4) << 2;    // lane-group dword offset in A chunk

    const unsigned* arow =
        (const unsigned*)(Ain + (mt * 16 + (lane & 15)) * HIDP);
    const _Float16* wp = Wp + ((nt0 << 9) | (lane << 4));  // one pointer; j,kt via immediates/increment

    v8f acc[7];
#pragma unroll
    for (int j = 0; j < 7; ++j) acc[j] = (v8f){};

#pragma unroll 1
    for (int kt = 0; kt < ktiles; ++kt) {
        v16h a = load_a_frag(arow, kt, g4);
#pragma unroll
        for (int j = 0; j < 7; ++j) {
            v16h b = *(const v16h*)(wp + (j << 9));   // offset j*1024B folded as immediate
            acc[j] = __builtin_amdgcn_wmma_f32_16x16x32_f16(false, a, false, b,
                                                            (short)0, acc[j], false, false);
        }
        wp += ntiles << 9;                // advance one k-chunk of packed weights
    }

#pragma unroll
    for (int j = 0; j < 7; ++j) {
        int n = ((nt0 + j) << 4) + (lane & 15);
        float bv = biasp[n];             // zero-padded: no guard, EXEC stays full
#pragma unroll
        for (int v = 0; v < 8; ++v) {
            float x = fmaxf(acc[j][v] + bv, 0.0f);
            int row = (mt << 4) + ((lane >> 4) << 3) + v;
            Aout[row * HIDP + n] = (_Float16)x;
        }
    }
    __syncthreads();
}

// ---------------------------------------------------------------------------
// Output layer fused with SDE update (4 n-tiles per wave, no divergence:
// padded columns have u == 0 exactly, so their updates are no-ops and the
// dW read stays in-bounds).
//   u = Ain @ Wout + bout ;  err += u*u*h ;  S += u*h + u*sqrt(h)*dW[b,t,d]
// ---------------------------------------------------------------------------
__device__ __forceinline__ void layer_out(const _Float16* Ain,
                                          const _Float16* __restrict__ Wp,
                                          const float* __restrict__ boutp,
                                          const float* __restrict__ dW,
                                          float* Ssh, float* Esh,
                                          int row0, int step, float h, float sqh,
                                          int lane, int wave) {
    const int ntiles = 8;                // OUTP / 16
    const int ktiles = 7;                // HIDP / 32
    const int mt  = wave >> 1;
    const int nt0 = (wave & 1) * 4;
    const int g4  = (lane >> 4) << 2;

    const unsigned* arow =
        (const unsigned*)(Ain + (mt * 16 + (lane & 15)) * HIDP);
    const _Float16* wp = Wp + ((nt0 << 9) | (lane << 4));

    v8f acc[4];
#pragma unroll
    for (int j = 0; j < 4; ++j) acc[j] = (v8f){};

#pragma unroll 1
    for (int kt = 0; kt < ktiles; ++kt) {
        v16h a = load_a_frag(arow, kt, g4);
#pragma unroll
        for (int j = 0; j < 4; ++j) {
            v16h b = *(const v16h*)(wp + (j << 9));
            acc[j] = __builtin_amdgcn_wmma_f32_16x16x32_f16(false, a, false, b,
                                                            (short)0, acc[j], false, false);
        }
        wp += ntiles << 9;
    }

#pragma unroll
    for (int j = 0; j < 4; ++j) {
        int n = ((nt0 + j) << 4) + (lane & 15);
        float bv = boutp[n];
#pragma unroll
        for (int v = 0; v < 8; ++v) {
            int row = (mt << 4) + ((lane >> 4) << 3) + v;
            float u = acc[j][v] + bv;
            float dw = dW[(size_t)(row0 + row) * (DIM * T_) + step * DIM + n];
            Esh[(row << 7) + n] += u * u * h;
            Ssh[(row << 7) + n] += u * h + u * sqh * dw;
        }
    }
    __syncthreads();
}

// ---------------------------------------------------------------------------
// Main persistent kernel: one block owns 32 batch rows for all 100 steps.
// ---------------------------------------------------------------------------
__global__ __launch_bounds__(BLOCK) void sde_mlp_kernel(
    const float* __restrict__ S0, const float* __restrict__ tg,
    const float* __restrict__ dW,
    const float* __restrict__ b0p, const float* __restrict__ b1p,
    const float* __restrict__ b2p, const float* __restrict__ b3p,
    const float* __restrict__ boutp,
    const _Float16* __restrict__ W0p, const _Float16* __restrict__ W1p,
    const _Float16* __restrict__ W2p, const _Float16* __restrict__ W3p,
    const _Float16* __restrict__ Wop,
    float* __restrict__ out) {
    __shared__ _Float16 A0[MTILE * HIDP];   // 14 KB
    __shared__ _Float16 A1[MTILE * HIDP];   // 14 KB
    __shared__ float    Ssh[MTILE * 128];   // 16 KB (cols 0..99 valid)
    __shared__ float    Esh[MTILE * 128];   // 16 KB

    const int tid  = threadIdx.x;
    const int lane = tid & 31;
    const int wave = tid >> 5;
    const int row0 = blockIdx.x * MTILE;

    const float h   = tg[1] - tg[0];
    const float sqh = sqrtf(h);

    // Init state / error accumulators
    for (int i = tid; i < MTILE * 128; i += BLOCK) {
        int r = i >> 7, c = i & 127;
        Ssh[i] = (c < DIM) ? S0[(size_t)(row0 + r) * DIM + c] : 0.0f;
        Esh[i] = 0.0f;
    }
    __syncthreads();

#pragma unroll 1
    for (int step = 0; step < T_ - 1; ++step) {
        // Prefetch this step's dW slice (only HBM stream in the loop):
        // 32 rows x 400 B; one cacheline-granular prefetch per thread.
        {
            int r   = tid >> 2;       // 0..31
            int seg = tid & 3;        // 4 x 128 B segments per row
            __builtin_prefetch(&dW[(size_t)(row0 + r) * (DIM * T_) + step * DIM + seg * 32],
                               0, 3);
        }

        // Build f16 input activations [t, S, 0-pad] into A0 (first 128 cols)
        float tcur = tg[step];
        for (int i = tid; i < MTILE * INP; i += BLOCK) {
            int r = i >> 7, c = i & 127;
            float v = (c == 0) ? tcur : ((c <= DIM) ? Ssh[(r << 7) + (c - 1)] : 0.0f);
            A0[r * HIDP + c] = (_Float16)v;
        }
        __syncthreads();

        layer_hidden(A0, A1, W0p, /*ktiles=*/4, b0p, lane, wave);
        layer_hidden(A1, A0, W1p, /*ktiles=*/7, b1p, lane, wave);
        layer_hidden(A0, A1, W2p, /*ktiles=*/7, b2p, lane, wave);
        layer_hidden(A1, A0, W3p, /*ktiles=*/7, b3p, lane, wave);
        layer_out(A0, Wop, boutp, dW, Ssh, Esh, row0, step, h, sqh, lane, wave);
    }

    // out = err + S_final^2
    for (int i = tid; i < MTILE * DIM; i += BLOCK) {
        int r = i / DIM, d = i - r * DIM;
        float s = Ssh[(r << 7) + d];
        out[(size_t)(row0 + r) * DIM + d] = Esh[(r << 7) + d] + s * s;
    }
}

extern "C" void kernel_launch(void* const* d_in, const int* in_sizes, int n_in,
                              void* d_out, int out_size, void* d_ws, size_t ws_size,
                              hipStream_t stream) {
    (void)in_sizes; (void)n_in; (void)out_size; (void)ws_size;

    const float* S0   = (const float*)d_in[0];
    const float* tg   = (const float*)d_in[1];
    const float* dW   = (const float*)d_in[2];
    const float* W0   = (const float*)d_in[3];
    const float* b0   = (const float*)d_in[4];
    const float* W1   = (const float*)d_in[5];
    const float* b1   = (const float*)d_in[6];
    const float* W2   = (const float*)d_in[7];
    const float* b2   = (const float*)d_in[8];
    const float* W3   = (const float*)d_in[9];
    const float* b3   = (const float*)d_in[10];
    const float* Wout = (const float*)d_in[11];
    const float* bout = (const float*)d_in[12];
    float* out = (float*)d_out;

    _Float16* ws  = (_Float16*)d_ws;
    _Float16* W0p = ws;
    _Float16* W1p = W0p + SZ_W0;
    _Float16* W2p = W1p + SZ_WH;
    _Float16* W3p = W2p + SZ_WH;
    _Float16* Wop = W3p + SZ_WH;

    float* bpad = (float*)(ws + TOTAL_W);   // 415744 B offset, 4B-aligned
    float* b0p  = bpad;
    float* b1p  = b0p + HIDP;
    float* b2p  = b1p + HIDP;
    float* b3p  = b2p + HIDP;
    float* bop  = b3p + HIDP;               // 128 floats

    // Pack weights into WMMA-fragment-major f16 layout (L2-resident afterwards)
    pack_weights<<<(SZ_W0 + 255) / 256, 256, 0, stream>>>(W0,   W0p, DIM + 1, WIDTH, 4, 14);
    pack_weights<<<(SZ_WH + 255) / 256, 256, 0, stream>>>(W1,   W1p, WIDTH,   WIDTH, 7, 14);
    pack_weights<<<(SZ_WH + 255) / 256, 256, 0, stream>>>(W2,   W2p, WIDTH,   WIDTH, 7, 14);
    pack_weights<<<(SZ_WH + 255) / 256, 256, 0, stream>>>(W3,   W3p, WIDTH,   WIDTH, 7, 14);
    pack_weights<<<(SZ_WO + 255) / 256, 256, 0, stream>>>(Wout, Wop, WIDTH,   DIM,   7, 8);

    // Zero-padded biases
    pack_bias<<<1, 256, 0, stream>>>(b0,   b0p, WIDTH, HIDP);
    pack_bias<<<1, 256, 0, stream>>>(b1,   b1p, WIDTH, HIDP);
    pack_bias<<<1, 256, 0, stream>>>(b2,   b2p, WIDTH, HIDP);
    pack_bias<<<1, 256, 0, stream>>>(b3,   b3p, WIDTH, HIDP);
    pack_bias<<<1, 256, 0, stream>>>(bout, bop, DIM,   OUTP);

    // Persistent time-stepping kernel: 128 blocks x 32 rows = 4096 batch rows
    sde_mlp_kernel<<<BATCH / MTILE, BLOCK, 0, stream>>>(
        S0, tg, dW, b0p, b1p, b2p, b3p, bop, W0p, W1p, W2p, W3p, Wop, out);
}
